// KMGCN_47863115547260
// MI455X (gfx1250) — compile-verified, matrix operand order checked by reading, not run on hardware
//
#include <hip/hip_runtime.h>

// ---------------------------------------------------------------------------
// GCN inference for MI455X (gfx1250, wave32).
// Memory-bound (~1.3 GB traffic -> ~55us roofline @ 23.3 TB/s). Dense
// feature transforms: v_wmma_f32_16x16x32_f16, one wave per 16-row tile
// computing all 6 column tiles (A fragment reused 6x, B pre-packed into the
// exact per-lane fragment layout so every load is a contiguous b128).
// ---------------------------------------------------------------------------

typedef __attribute__((ext_vector_type(16))) _Float16 v16h;
typedef __attribute__((ext_vector_type(8)))  float    v8f;

#define N_NODES 50000
#define N_EDGES 800000
#define D_IN    300
#define D_H     96
#define D_OUT   2
#define G_GR    64
#define TILES_N (D_H / 16)          // 6 column tiles

// ---------------- utility kernels ----------------

__global__ void KMGCN_zero_f32(float* p, int n) {
    int i = blockIdx.x * blockDim.x + threadIdx.x;
    if (i < n) p[i] = 0.0f;
}

__global__ void KMGCN_zero_i32(int* p, int n) {
    int i = blockIdx.x * blockDim.x + threadIdx.x;
    if (i < n) p[i] = 0;
}

__global__ void KMGCN_deg_count(const int* __restrict__ dst, int* __restrict__ cnt) {
    int e = blockIdx.x * blockDim.x + threadIdx.x;
    if (e < N_EDGES) atomicAdd(&cnt[dst[e]], 1);
}

__global__ void KMGCN_compute_dis(const int* __restrict__ cnt, float* __restrict__ dis) {
    int n = blockIdx.x * blockDim.x + threadIdx.x;
    if (n < N_NODES) dis[n] = rsqrtf((float)cnt[n] + 1.0f);   // deg includes self-loop
}

// ---------------- W pre-pack into WMMA B-fragment layout ----------------
// CDNA5 16-bit B 32x16 fragment (mirror of the documented A layout):
//   lane: col = lane&15, khalf = (lane>>4)*8
//   half j = 2v+hi -> K = khalf + ((v<4) ? 2v : 8+2v) + hi
// Packed as Bpack[(chunk*TILES_N + tile)*512 + lane*16 + j], zero-padded K.
template<int K>
__global__ void KMGCN_pack_w(const float* __restrict__ W, _Float16* __restrict__ Bpack) {
    constexpr int NCHUNK = (K + 31) / 32;
    const int idx = blockIdx.x * blockDim.x + threadIdx.x;
    if (idx >= NCHUNK * TILES_N * 512) return;
    const int c    = idx / (TILES_N * 512);
    const int rem  = idx % (TILES_N * 512);
    const int t    = rem / 512;
    const int li   = rem % 512;
    const int lane = li / 16;
    const int j    = li % 16;
    const int v    = j >> 1;
    const int hi   = j & 1;
    const int khalf = (lane >> 4) << 3;
    const int ko    = (v < 4) ? (2 * v) : (8 + 2 * v);
    const int kk    = c * 32 + khalf + ko + hi;
    const int col   = t * 16 + (lane & 15);
    Bpack[idx] = (kk < K) ? (_Float16)W[kk * D_H + col] : (_Float16)0.0f;
}

// ---------------- WMMA GEMM: C[N_NODES x 96] = A[N_NODES x K] * W[K x 96] ----
// One wave32 per 16-row tile; 6 accumulators cover all 96 output columns.
// A fragment per chunk: two contiguous 16B pairs (k 0..7 and 16..23 relative
// to this lane's khalf) -> 4x global_load_b128, converted f32->f16 in regs.
// B fragment: single contiguous 32B load from pre-packed Bpack.
template<int K>
__global__ __launch_bounds__(256) void KMGCN_gemm_wmma(
    const float* __restrict__ A, const _Float16* __restrict__ Bpack,
    float* __restrict__ C)
{
    const int wave = blockIdx.x * (blockDim.x >> 5) + (threadIdx.x >> 5);
    const int lane = threadIdx.x & 31;
    constexpr int TILES_M = N_NODES / 16;
    if (wave >= TILES_M) return;

    const int l15   = lane & 15;
    const int khalf = (lane >> 4) << 3;                 // 0 | 8
    const float* Arow = A + (long)(wave * 16 + l15) * K;
    const _Float16* Bl = Bpack + lane * 16;

    v8f acc[TILES_N] = {};

    constexpr int FULL = K / 32;
    for (int c = 0; c < FULL; ++c) {
        const int kb = c * 32 + khalf;
        const float4 a01 = *(const float4*)(Arow + kb);
        const float4 a23 = *(const float4*)(Arow + kb + 4);
        const float4 a45 = *(const float4*)(Arow + kb + 16);
        const float4 a67 = *(const float4*)(Arow + kb + 20);
        v16h a;
        a[0]  = (_Float16)a01.x;  a[1]  = (_Float16)a01.y;
        a[2]  = (_Float16)a01.z;  a[3]  = (_Float16)a01.w;
        a[4]  = (_Float16)a23.x;  a[5]  = (_Float16)a23.y;
        a[6]  = (_Float16)a23.z;  a[7]  = (_Float16)a23.w;
        a[8]  = (_Float16)a45.x;  a[9]  = (_Float16)a45.y;
        a[10] = (_Float16)a45.z;  a[11] = (_Float16)a45.w;
        a[12] = (_Float16)a67.x;  a[13] = (_Float16)a67.y;
        a[14] = (_Float16)a67.z;  a[15] = (_Float16)a67.w;

        const _Float16* bp = Bl + (c * TILES_N) * 512;
#pragma unroll
        for (int t = 0; t < TILES_N; ++t) {
            const v16h b = *(const v16h*)(bp + t * 512);
            acc[t] = __builtin_amdgcn_wmma_f32_16x16x32_f16(
                false, a, false, b, (short)0, acc[t], false, false);
        }
    }

    if constexpr ((K % 32) != 0) {                       // K=300 tail (12 k's)
        constexpr int c  = FULL;
        const int kb = c * 32 + khalf;
        v16h a = {};
#pragma unroll
        for (int v = 0; v < 8; ++v) {
            const int ko = (v < 4) ? (2 * v) : (8 + 2 * v);
            const int kk = kb + ko;                       // even; K even
            if (kk < K) {
                const float2 av = *(const float2*)(Arow + kk);
                a[2 * v]     = (_Float16)av.x;
                a[2 * v + 1] = (_Float16)av.y;
            }
        }
        const _Float16* bp = Bl + (c * TILES_N) * 512;    // zero-padded, no guard
#pragma unroll
        for (int t = 0; t < TILES_N; ++t) {
            const v16h b = *(const v16h*)(bp + t * 512);
            acc[t] = __builtin_amdgcn_wmma_f32_16x16x32_f16(
                false, a, false, b, (short)0, acc[t], false, false);
        }
    }

    // C/D layout: VGPR r -> row khalf+r, col = lane&15 (+16 per column tile)
    float* Crow = C + (long)(wave * 16 + khalf) * D_H + l15;
#pragma unroll
    for (int t = 0; t < TILES_N; ++t)
#pragma unroll
        for (int r = 0; r < 8; ++r)
            Crow[r * D_H + t * 16] = acc[t][r];
}

// ---------------- propagation ----------------

// agg = h * dis^2  (self-loop term; full overwrite => replay-safe init)
__global__ void KMGCN_self_init(const float* __restrict__ h, const float* __restrict__ dis,
                                float* __restrict__ agg) {
    int i = blockIdx.x * blockDim.x + threadIdx.x;
    if (i < N_NODES * D_H) {
        float d = dis[i / D_H];
        agg[i] = h[i] * d * d;
    }
}

// agg[dst] += h[src] * dis[src]*dis[dst]; thread = (edge, quad of 4 feats)
__global__ void KMGCN_edge_scatter(const float* __restrict__ h, const float* __restrict__ dis,
                                   const int* __restrict__ src, const int* __restrict__ dst,
                                   float* __restrict__ agg) {
    int idx = blockIdx.x * blockDim.x + threadIdx.x;
    if (idx >= N_EDGES * (D_H / 4)) return;
    const int e = idx / (D_H / 4);
    const int q = idx % (D_H / 4);
    const int s = src[e];
    const int d = dst[e];
    const float norm = dis[s] * dis[d];
    const float4 hv = *(const float4*)(h + (long)s * D_H + q * 4);
    float* ap = agg + (long)d * D_H + q * 4;
    atomicAdd(ap + 0, hv.x * norm);
    atomicAdd(ap + 1, hv.y * norm);
    atomicAdd(ap + 2, hv.z * norm);
    atomicAdd(ap + 3, hv.w * norm);
}

__global__ void KMGCN_bias_relu(float* __restrict__ agg, const float* __restrict__ b) {
    int i = blockIdx.x * blockDim.x + threadIdx.x;
    if (i < N_NODES * D_H) {
        float v = agg[i] + b[i % D_H];
        agg[i] = v > 0.0f ? v : 0.0f;
    }
}

// ---------------- pooling + FC ----------------

__global__ void KMGCN_pool_sum(const float* __restrict__ h, const int* __restrict__ batch,
                               float* __restrict__ pooled) {
    int idx = blockIdx.x * blockDim.x + threadIdx.x;
    if (idx >= N_NODES * (D_H / 4)) return;
    const int n = idx / (D_H / 4);
    const int q = idx % (D_H / 4);
    const int g = batch[n];
    const float4 hv = *(const float4*)(h + (long)n * D_H + q * 4);
    float* pp = pooled + (long)g * D_H + q * 4;
    atomicAdd(pp + 0, hv.x);
    atomicAdd(pp + 1, hv.y);
    atomicAdd(pp + 2, hv.z);
    atomicAdd(pp + 3, hv.w);
}

__global__ void KMGCN_pool_cnt(const int* __restrict__ batch, float* __restrict__ counts) {
    int n = blockIdx.x * blockDim.x + threadIdx.x;
    if (n < N_NODES) atomicAdd(&counts[batch[n]], 1.0f);
}

__global__ void KMGCN_fc(const float* __restrict__ pooled, const float* __restrict__ counts,
                         const float* __restrict__ Wfc, const float* __restrict__ bfc,
                         float* __restrict__ out) {
    int i = threadIdx.x;
    if (i >= G_GR * D_OUT) return;
    const int g = i / D_OUT;
    const int o = i % D_OUT;
    const float inv = 1.0f / fmaxf(counts[g], 1.0f);
    float s = bfc[o];
#pragma unroll 8
    for (int k = 0; k < D_H; ++k)
        s += pooled[g * D_H + k] * inv * Wfc[k * D_OUT + o];
    out[i] = s;
}

// ---------------- launch ----------------

static inline char* ws_take(char*& p, size_t bytes) {
    char* r = p;
    p += (bytes + 255) & ~(size_t)255;
    return r;
}

extern "C" void kernel_launch(void* const* d_in, const int* in_sizes, int n_in,
                              void* d_out, int out_size, void* d_ws, size_t ws_size,
                              hipStream_t stream) {
    (void)in_sizes; (void)n_in; (void)out_size; (void)ws_size;

    const float* x    = (const float*)d_in[0];
    const float* W1   = (const float*)d_in[1];
    const float* b1   = (const float*)d_in[2];
    const float* W2   = (const float*)d_in[3];
    const float* b2   = (const float*)d_in[4];
    const float* Wfc  = (const float*)d_in[5];
    const float* bfc  = (const float*)d_in[6];
    const int*   src  = (const int*)d_in[7];
    const int*   dst  = (const int*)d_in[8];
    const int*   batch= (const int*)d_in[9];
    float* out        = (float*)d_out;

    constexpr int NCH1 = (D_IN + 31) / 32;   // 10 chunks for K=300
    constexpr int NCH2 = (D_H + 31) / 32;    // 3 chunks for K=96

    char* p = (char*)d_ws;
    int*      cnt    = (int*)     ws_take(p, (size_t)N_NODES * 4);
    float*    dis    = (float*)   ws_take(p, (size_t)N_NODES * 4);
    float*    hbuf   = (float*)   ws_take(p, (size_t)N_NODES * D_H * 4);
    float*    agg    = (float*)   ws_take(p, (size_t)N_NODES * D_H * 4);
    float*    pooled = (float*)   ws_take(p, (size_t)G_GR * D_H * 4);
    float*    counts = (float*)   ws_take(p, (size_t)G_GR * 4);
    _Float16* w1p    = (_Float16*)ws_take(p, (size_t)NCH1 * TILES_N * 512 * 2);
    _Float16* w2p    = (_Float16*)ws_take(p, (size_t)NCH2 * TILES_N * 512 * 2);

    const int T = 256;
    auto blk = [](int n, int t) { return (n + t - 1) / t; };

    // --- degrees / normalization ---
    KMGCN_zero_i32<<<blk(N_NODES, T), T, 0, stream>>>(cnt, N_NODES);
    KMGCN_zero_f32<<<blk(G_GR * D_H, T), T, 0, stream>>>(pooled, G_GR * D_H);
    KMGCN_zero_f32<<<1, T, 0, stream>>>(counts, G_GR);
    KMGCN_deg_count<<<blk(N_EDGES, T), T, 0, stream>>>(dst, cnt);
    KMGCN_compute_dis<<<blk(N_NODES, T), T, 0, stream>>>(cnt, dis);

    // --- weight pre-pack into B-fragment layout (tiny; L2-resident) ---
    KMGCN_pack_w<D_IN><<<blk(NCH1 * TILES_N * 512, T), T, 0, stream>>>(W1, w1p);
    KMGCN_pack_w<D_H ><<<blk(NCH2 * TILES_N * 512, T), T, 0, stream>>>(W2, w2p);

    const int gemmBlk = blk(N_NODES / 16, T / 32);   // 3125 waves, 8 per block

    // --- layer 1: h = x @ W1 ; propagate ; relu(+b1) ---
    KMGCN_gemm_wmma<D_IN><<<gemmBlk, T, 0, stream>>>(x, w1p, hbuf);
    KMGCN_self_init<<<blk(N_NODES * D_H, T), T, 0, stream>>>(hbuf, dis, agg);
    KMGCN_edge_scatter<<<blk(N_EDGES * (D_H / 4), T), T, 0, stream>>>(hbuf, dis, src, dst, agg);
    KMGCN_bias_relu<<<blk(N_NODES * D_H, T), T, 0, stream>>>(agg, b1);

    // --- layer 2: h = relu1 @ W2 ; propagate ; relu(+b2) ---
    KMGCN_gemm_wmma<D_H><<<gemmBlk, T, 0, stream>>>(agg, w2p, hbuf);
    KMGCN_self_init<<<blk(N_NODES * D_H, T), T, 0, stream>>>(hbuf, dis, agg);
    KMGCN_edge_scatter<<<blk(N_EDGES * (D_H / 4), T), T, 0, stream>>>(hbuf, dis, src, dst, agg);
    KMGCN_bias_relu<<<blk(N_NODES * D_H, T), T, 0, stream>>>(agg, b2);

    // --- global mean pool + FC ---
    KMGCN_pool_sum<<<blk(N_NODES * (D_H / 4), T), T, 0, stream>>>(agg, batch, pooled);
    KMGCN_pool_cnt<<<blk(N_NODES, T), T, 0, stream>>>(batch, counts);
    KMGCN_fc<<<1, 128, 0, stream>>>(pooled, counts, Wfc, bfc, out);
}
